// GIN_71708773974793
// MI455X (gfx1250) — matrix operand order repaired and checked by reading, not compiled, not run on hardware
//
#include <hip/hip_runtime.h>

#define N_NODES  50000
#define N_EDGES  800000
#define DIM      128
#define OUT_DIM  10
#define N_GRAPHS 100
#define BN_EPS   1e-5f

typedef __attribute__((ext_vector_type(16))) __bf16 v16bf;
typedef __attribute__((ext_vector_type(8)))  __bf16 v8bf;
typedef __attribute__((ext_vector_type(8)))  float  v8f;

__device__ __forceinline__ unsigned short f2bfr(float f) {
  union { float f; unsigned u; } v; v.f = f;
  unsigned u = v.u;
  return (unsigned short)((u + 0x7FFFu + ((u >> 16) & 1u)) >> 16);  // RNE
}

__device__ __forceinline__ __bf16 f2bf(float f) {
  return __builtin_bit_cast(__bf16, f2bfr(f));
}

// ---------------------------------------------------------------- utilities
__global__ void zero_f32(float* __restrict__ p, int n) {
  int i = blockIdx.x * blockDim.x + threadIdx.x;
  if (i < n) p[i] = 0.0f;
}

__global__ void copy_f32v4(float* __restrict__ d, const float* __restrict__ s, int n4) {
  int i = blockIdx.x * blockDim.x + threadIdx.x;
  if (i < n4) reinterpret_cast<float4*>(d)[i] = reinterpret_cast<const float4*>(s)[i];
}

// W [K=128][N=128] f32 row-major  ->  Wt [N][K] bf16 (B-fragment friendly)
__global__ void wt_prep(const float* __restrict__ W, __bf16* __restrict__ Wt) {
  int idx = blockIdx.x * blockDim.x + threadIdx.x;   // 0..16383
  int k = idx >> 7, n = idx & 127;
  Wt[n * DIM + k] = f2bf(W[idx]);
}

// ---------------------------------------------------- edge scatter: wave/edge
__global__ __launch_bounds__(256) void scatter_add(const float* __restrict__ x,
                                                   const int* __restrict__ src,
                                                   const int* __restrict__ dst,
                                                   float* __restrict__ agg) {
  int e = blockIdx.x * 8 + (threadIdx.x >> 5);
  if (e >= N_EDGES) return;
  int lane = threadIdx.x & 31;
  int s = src[e], d = dst[e];
  float4 v = reinterpret_cast<const float4*>(x)[(size_t)s * 32 + lane];
  float* o = agg + (size_t)d * DIM + lane * 4;
  atomicAdd(o + 0, v.x); atomicAdd(o + 1, v.y);
  atomicAdd(o + 2, v.z); atomicAdd(o + 3, v.w);
}

// ---------------------------------------------------- graph pooling: wave/node
__global__ __launch_bounds__(256) void pool_add(const float* __restrict__ h,
                                                const int* __restrict__ batch,
                                                float* __restrict__ hg) {
  int node = blockIdx.x * 8 + (threadIdx.x >> 5);
  if (node >= N_NODES) return;
  int lane = threadIdx.x & 31;
  int g = batch[node];
  float4 v = reinterpret_cast<const float4*>(h)[(size_t)node * 32 + lane];
  float* o = hg + (size_t)g * DIM + lane * 4;
  atomicAdd(o + 0, v.x); atomicAdd(o + 1, v.y);
  atomicAdd(o + 2, v.z); atomicAdd(o + 3, v.w);
}

// ------------------------------------------------------------- WMMA GEMM
// out[N_NODES,128] = preop(A)[N_NODES,128] @ W[128,128] + bias
//   preop (if pre_scale): a = relu(a*scale[k] + shift[k])   (fused BN+ReLU)
//   optional ReLU on output, optional column sum/sumsq accumulation (for BN).
// Block: 128 threads (4 waves) -> one 16-row M tile; wave w -> N tiles 2w, 2w+1.
// A tile staged in LDS as bf16 (converted ONCE during staging), padded stride
// 136 bf16 (272 B = 68 words -> lane m lands on bank group 4m mod 64: conflict-free).
#define LDS_STRIDE 136

__global__ __launch_bounds__(128) void gemm128_wmma(
    const float* __restrict__ A, const __bf16* __restrict__ Wt,
    const float* __restrict__ bias,
    const float* __restrict__ pre_scale, const float* __restrict__ pre_shift,
    float* __restrict__ outp,
    float* __restrict__ col_sum, float* __restrict__ col_sumsq,
    int relu_out)
{
  __shared__ __align__(16) unsigned short ldsb[16 * LDS_STRIDE];
  const int tid   = threadIdx.x;
  const int mbase = blockIdx.x * 16;

  // ---- stage 16x128 A tile into LDS as bf16, with optional BN+ReLU pre-op
  {
    const int row = tid >> 3;
    const int cs  = (tid & 7) * 16;
    const float* g = A + (size_t)(mbase + row) * DIM + cs;
    const bool pre = (pre_scale != nullptr);
    float vals[16];
#pragma unroll
    for (int i = 0; i < 16; i += 4) {
      float4 v = *reinterpret_cast<const float4*>(g + i);
      if (pre) {
        v.x = fmaxf(v.x * pre_scale[cs + i + 0] + pre_shift[cs + i + 0], 0.0f);
        v.y = fmaxf(v.y * pre_scale[cs + i + 1] + pre_shift[cs + i + 1], 0.0f);
        v.z = fmaxf(v.z * pre_scale[cs + i + 2] + pre_shift[cs + i + 2], 0.0f);
        v.w = fmaxf(v.w * pre_scale[cs + i + 3] + pre_shift[cs + i + 3], 0.0f);
      }
      vals[i + 0] = v.x; vals[i + 1] = v.y; vals[i + 2] = v.z; vals[i + 3] = v.w;
    }
    unsigned q[8];
#pragma unroll
    for (int i = 0; i < 8; ++i)
      q[i] = (unsigned)f2bfr(vals[2 * i]) | ((unsigned)f2bfr(vals[2 * i + 1]) << 16);
    uint4* l = reinterpret_cast<uint4*>(&ldsb[row * LDS_STRIDE + cs]);  // 16B aligned
    l[0] = make_uint4(q[0], q[1], q[2], q[3]);
    l[1] = make_uint4(q[4], q[5], q[6], q[7]);
  }
  __syncthreads();

  const int lane = tid & 31;
  const int wid  = tid >> 5;
  const int n0   = wid * 2;            // first of two N tiles for this wave
  const int lm   = lane & 15;
  const int hi   = (lane & 16) ? 1 : 0;

  const float bv0 = bias[n0 * 16 + lm];
  const float bv1 = bias[(n0 + 1) * 16 + lm];
  v8f c0, c1;
#pragma unroll
  for (int r = 0; r < 8; ++r) { c0[r] = bv0; c1[r] = bv1; }

#pragma unroll
  for (int ks = 0; ks < 4; ++ks) {
    // A fragment (lanes 0-15 -> K 0-15 of step; lanes 16-31 -> K 16-31; matches B)
    const int kb = ks * 32 + (hi ? 16 : 0);
    const unsigned short* lp = &ldsb[lm * LDS_STRIDE + kb];
    v8bf alo = *reinterpret_cast<const v8bf*>(lp);        // K kb..kb+7   (ds_load_b128)
    v8bf ahi = *reinterpret_cast<const v8bf*>(lp + 8);    // K kb+8..kb+15
    v16bf a = __builtin_shufflevector(alo, ahi, 0, 1, 2, 3, 4, 5, 6, 7,
                                      8, 9, 10, 11, 12, 13, 14, 15);

    // B fragments (lanes 0-15 -> K 0-15 of step; lanes 16-31 -> K 16-31; N = lane%16)
    const int krow = ks * 32 + (hi ? 16 : 0);
    v16bf bf0 = *reinterpret_cast<const v16bf*>(Wt + (size_t)(n0 * 16 + lm) * DIM + krow);
    v16bf bf1 = *reinterpret_cast<const v16bf*>(Wt + (size_t)((n0 + 1) * 16 + lm) * DIM + krow);

    c0 = __builtin_amdgcn_wmma_f32_16x16x32_bf16(false, a, false, bf0, (short)0, c0, false, false);
    c1 = __builtin_amdgcn_wmma_f32_16x16x32_bf16(false, a, false, bf1, (short)0, c1, false, false);
  }

  // ---- epilogue: store (+ optional ReLU) and accumulate BN column stats
  const int mrow = mbase + hi * 8;
  const int col0 = n0 * 16 + lm;
  const int col1 = col0 + 16;
  float s0 = 0.f, q0 = 0.f, s1 = 0.f, q1 = 0.f;
#pragma unroll
  for (int r = 0; r < 8; ++r) {
    float v0 = c0[r], v1 = c1[r];
    if (relu_out) { v0 = fmaxf(v0, 0.f); v1 = fmaxf(v1, 0.f); }
    outp[(size_t)(mrow + r) * DIM + col0] = v0;
    outp[(size_t)(mrow + r) * DIM + col1] = v1;
    s0 += v0; q0 += v0 * v0; s1 += v1; q1 += v1 * v1;
  }
  if (col_sum) {
    atomicAdd(&col_sum[col0], s0);  atomicAdd(&col_sumsq[col0], q0);
    atomicAdd(&col_sum[col1], s1);  atomicAdd(&col_sumsq[col1], q1);
  }
}

// ---------------------------------------------------------- BN stat finalize
__global__ void bn_finalize(const float* __restrict__ sum, const float* __restrict__ sumsq,
                            const float* __restrict__ gamma, const float* __restrict__ beta,
                            float* __restrict__ scale, float* __restrict__ shift) {
  int j = threadIdx.x;
  const float invN = 1.0f / (float)N_NODES;
  float mu  = sum[j] * invN;
  float var = sumsq[j] * invN - mu * mu;
  float inv = rsqrtf(var + BN_EPS);
  float sc  = gamma[j] * inv;
  scale[j] = sc;
  shift[j] = beta[j] - mu * sc;
}

// ------------------------------------------------------------- MLP head
__global__ __launch_bounds__(128) void head(const float* __restrict__ hg,
                                            const float* __restrict__ Wl1, const float* __restrict__ bl1,
                                            const float* __restrict__ Wl2, const float* __restrict__ bl2,
                                            float* __restrict__ out) {
  __shared__ float row[DIM];
  __shared__ float y[DIM];
  int g = blockIdx.x, j = threadIdx.x;
  row[j] = hg[(size_t)g * DIM + j];
  __syncthreads();
  float acc = bl1[j];
  for (int k = 0; k < DIM; ++k) acc += row[k] * Wl1[k * DIM + j];
  y[j] = fmaxf(acc, 0.0f);
  __syncthreads();
  if (j < OUT_DIM) {
    float o = bl2[j];
    for (int k = 0; k < DIM; ++k) o += y[k] * Wl2[k * OUT_DIM + j];
    out[(size_t)g * OUT_DIM + j] = o;
  }
}

// ---------------------------------------------------------------- driver
extern "C" void kernel_launch(void* const* d_in, const int* in_sizes, int n_in,
                              void* d_out, int out_size, void* d_ws, size_t ws_size,
                              hipStream_t stream) {
  (void)in_sizes; (void)n_in; (void)out_size; (void)ws_size;
  const float* x     = (const float*)d_in[0];
  const int*   ei    = (const int*)d_in[1];
  const int*   srcI  = ei;
  const int*   dstI  = ei + N_EDGES;
  const int*   batch = (const int*)d_in[2];
  const float *W1a = (const float*)d_in[3],  *b1a = (const float*)d_in[4];
  const float *g1a = (const float*)d_in[5],  *be1a = (const float*)d_in[6];
  const float *W2a = (const float*)d_in[7],  *b2a = (const float*)d_in[8];
  const float *W1b = (const float*)d_in[9],  *b1b = (const float*)d_in[10];
  const float *g1b = (const float*)d_in[11], *be1b = (const float*)d_in[12];
  const float *W2b = (const float*)d_in[13], *b2b = (const float*)d_in[14];
  const float *Wl1 = (const float*)d_in[15], *bl1 = (const float*)d_in[16];
  const float *Wl2 = (const float*)d_in[17], *bl2 = (const float*)d_in[18];

  char* ws = (char*)d_ws;
  size_t off = 0;
  auto alloc = [&](size_t bytes) -> void* {
    void* p = ws + off;
    off = (off + bytes + 255) & ~(size_t)255;
    return p;
  };
  float*  agg   = (float*)alloc((size_t)N_NODES * DIM * 4);
  float*  tbuf  = (float*)alloc((size_t)N_NODES * DIM * 4);
  float*  xbuf  = (float*)alloc((size_t)N_NODES * DIM * 4);
  __bf16* Wt1a  = (__bf16*)alloc(DIM * DIM * 2);
  __bf16* Wt2a  = (__bf16*)alloc(DIM * DIM * 2);
  __bf16* Wt1b  = (__bf16*)alloc(DIM * DIM * 2);
  __bf16* Wt2b  = (__bf16*)alloc(DIM * DIM * 2);
  float*  stats = (float*)alloc(4 * DIM * 4);
  float*  ssum  = stats;
  float*  ssq   = stats + DIM;
  float*  scal  = stats + 2 * DIM;
  float*  shft  = stats + 3 * DIM;
  float*  hg    = (float*)alloc((size_t)N_GRAPHS * DIM * 4);

  const int MTILES = N_NODES / 16;             // 3125
  const int COPY4  = (N_NODES * DIM) / 4;      // 1.6M float4
  const int COPYB  = COPY4 / 256;              // 6250

  // weight prep (bf16, transposed)
  wt_prep<<<64, 256, 0, stream>>>(W1a, Wt1a);
  wt_prep<<<64, 256, 0, stream>>>(W2a, Wt2a);
  wt_prep<<<64, 256, 0, stream>>>(W1b, Wt1b);
  wt_prep<<<64, 256, 0, stream>>>(W2b, Wt2b);

  // ---------- GIN layer A ----------
  copy_f32v4<<<COPYB, 256, 0, stream>>>(agg, x, COPY4);                    // agg = x
  scatter_add<<<N_EDGES / 8, 256, 0, stream>>>(x, srcI, dstI, agg);        // agg += Σ neighbors
  zero_f32<<<1, 256, 0, stream>>>(ssum, 2 * DIM);
  gemm128_wmma<<<MTILES, 128, 0, stream>>>(agg, Wt1a, b1a, nullptr, nullptr,
                                           tbuf, ssum, ssq, 0);            // t = agg@W1a+b1a (+stats)
  bn_finalize<<<1, DIM, 0, stream>>>(ssum, ssq, g1a, be1a, scal, shft);
  gemm128_wmma<<<MTILES, 128, 0, stream>>>(tbuf, Wt2a, b2a, scal, shft,
                                           xbuf, nullptr, nullptr, 1);     // x1 = relu(relu(BN(t))@W2a+b2a)

  // ---------- GIN layer B ----------
  copy_f32v4<<<COPYB, 256, 0, stream>>>(agg, xbuf, COPY4);
  scatter_add<<<N_EDGES / 8, 256, 0, stream>>>(xbuf, srcI, dstI, agg);
  zero_f32<<<1, 256, 0, stream>>>(ssum, 2 * DIM);
  gemm128_wmma<<<MTILES, 128, 0, stream>>>(agg, Wt1b, b1b, nullptr, nullptr,
                                           tbuf, ssum, ssq, 0);
  bn_finalize<<<1, DIM, 0, stream>>>(ssum, ssq, g1b, be1b, scal, shft);
  gemm128_wmma<<<MTILES, 128, 0, stream>>>(tbuf, Wt2b, b2b, scal, shft,
                                           xbuf, nullptr, nullptr, 1);     // x2

  // ---------- pooling + head ----------
  zero_f32<<<(N_GRAPHS * DIM + 255) / 256, 256, 0, stream>>>(hg, N_GRAPHS * DIM);
  pool_add<<<N_NODES / 8, 256, 0, stream>>>(xbuf, batch, hg);
  head<<<N_GRAPHS, 128, 0, stream>>>(hg, Wl1, bl1, Wl2, bl2, (float*)d_out);
}
